// ExtendedQSUP_31653908972055
// MI455X (gfx1250) — matrix operand order repaired and checked by reading, not compiled, not run on hardware
//
#include <hip/hip_runtime.h>
#include <hip/hip_bf16.h>
#include <math.h>

// ---------------------------------------------------------------------------
// Types for CDNA5 WMMA (wave32): bf16 16x16x32 -> f32 accumulate
// ---------------------------------------------------------------------------
typedef __attribute__((ext_vector_type(16))) __bf16 v16bf;
typedef __attribute__((ext_vector_type(8)))  __bf16 v8bf;
typedef __attribute__((ext_vector_type(8)))  float  v8f;

#define EPSV 1e-8f

__device__ __forceinline__ __bf16 f2bf(float f) {
  return static_cast<__bf16>(f);   // native f32->bf16 convert (RNE)
}

// ---------------------------------------------------------------------------
// f32 -> bf16 conversion with optional zero-padding of the destination tail
// ---------------------------------------------------------------------------
__global__ void cvt_f32_bf16(const float* __restrict__ src, __bf16* __restrict__ dst,
                             long n, long ndst) {
  long i = (long)blockIdx.x * blockDim.x + threadIdx.x;
  long stride = (long)gridDim.x * blockDim.x;
  for (; i < ndst; i += stride)
    dst[i] = (i < n) ? f2bf(src[i]) : (__bf16)0.0f;
}

// ---------------------------------------------------------------------------
// WMMA GEMM:  C[M,N] = A[M,K] * B[N,K]^T  (+bias, + epilogue variant)
//   Block tile 128x128x32, 256 threads = 8 waves (4 M-waves x 2 N-waves),
//   each wave owns 32x64 = 8 WMMA accum tiles.
//   Tiles are staged memory->LDS with GLOBAL_LOAD_ASYNC_TO_LDS_B128 into a
//   ping-pong double buffer; ASYNCcnt ordered with s_wait_asynccnt, one
//   workgroup barrier per K-iteration.
// EPI: 0 = bias + f32 store
//      1 = gate = sigmoid(v)+0.1, RMW supR/supI *= gate
//      2 = exact GELU -> bf16 store
//      3 = bias/store guarded to col < Ncols (padded-N classifier head)
// ---------------------------------------------------------------------------
constexpr int BM = 128, BN = 128, BK = 32;
constexpr int LDT = BK + 16;   // 48 bf16 = 96B row stride: 32B-aligned frags

__device__ __forceinline__ void async_b128_to_lds(const __bf16* gptr, const __bf16* lptr) {
  const unsigned lds_addr = (unsigned)(size_t)lptr;              // low 32b = LDS offset
  const unsigned long long gaddr = (unsigned long long)(size_t)gptr;
  asm volatile("global_load_async_to_lds_b128 %0, %1, off"
               :: "v"(lds_addr), "v"(gaddr) : "memory");
}
__device__ __forceinline__ void wait_async0() {
  asm volatile("s_wait_asynccnt 0x0" ::: "memory");
}

template<int EPI>
__global__ __launch_bounds__(256)
void gemm_wmma(const __bf16* __restrict__ Ab, const __bf16* __restrict__ Bb,
               float* __restrict__ Cf, __bf16* __restrict__ Cb,
               const float* __restrict__ bias,
               float* __restrict__ supR, float* __restrict__ supI,
               int M, int N, int K, int ldC, int Ncols)
{
  __shared__ __bf16 sA[2][BM * LDT];
  __shared__ __bf16 sB[2][BN * LDT];

  const int tid   = threadIdx.x;
  const int lane  = tid & 31;
  const int wave  = tid >> 5;
  const int waveM = wave >> 1;       // 0..3
  const int waveN = wave & 1;        // 0..1
  const int bM    = blockIdx.y * BM;
  const int bN    = blockIdx.x * BN;

  v8f acc[2][4];
  #pragma unroll
  for (int i = 0; i < 2; i++)
    #pragma unroll
    for (int j = 0; j < 4; j++)
      acc[i][j] = (v8f)(0.0f);

  const int lrow = tid >> 1;         // 0..127: tile row loaded by this thread
  const int half = tid & 1;          // which 16-wide K chunk
  const __bf16* aSrc = Ab + (size_t)(bM + lrow) * K + half * 16;
  const __bf16* bSrc = Bb + (size_t)(bN + lrow) * K + half * 16;
  const int ldsIdx = lrow * LDT + half * 16;

  // prologue: stage tile 0 into buffer 0
  async_b128_to_lds(aSrc, &sA[0][ldsIdx]);
  async_b128_to_lds(bSrc, &sB[0][ldsIdx]);
  wait_async0();
  __syncthreads();

  const int nIter = K / BK;
  for (int it = 0; it < nIter; ++it) {
    const int buf = it & 1;
    if (it + 1 < nIter) {            // overlap: async-stage next tile
      const int k1 = (it + 1) * BK;
      async_b128_to_lds(aSrc + k1, &sA[buf ^ 1][ldsIdx]);
      async_b128_to_lds(bSrc + k1, &sB[buf ^ 1][ldsIdx]);
    }

    // Fragment loads in contiguous k order (same for A and B):
    //  A 16x32 bf16: lane<16 holds K 0..15, lane>=16 holds K 16..31 (contiguous)
    //  B 32x16 bf16: lane<16 holds K 0..15, lane>=16 holds K 16..31 (contiguous)
    const int lr = lane & 15;
    const int kA = (lane >> 4) * 16;
    const int kB = (lane >> 4) * 16;
    #pragma unroll
    for (int mt = 0; mt < 2; mt++) {
      const __bf16* ap = &sA[buf][(waveM * 32 + mt * 16 + lr) * LDT];
      v8bf a0 = *(const v8bf*)(ap + kA);
      v8bf a1 = *(const v8bf*)(ap + kA + 8);
      v16bf a = __builtin_shufflevector(a0, a1, 0,1,2,3,4,5,6,7,
                                               8,9,10,11,12,13,14,15);
      #pragma unroll
      for (int nt = 0; nt < 4; nt++) {
        v16bf b = *(const v16bf*)(&sB[buf][(waveN * 64 + nt * 16 + lr) * LDT + kB]);
        acc[mt][nt] = __builtin_amdgcn_wmma_f32_16x16x32_bf16(
            false, a, false, b, (short)0, acc[mt][nt], false, false);
      }
    }

    wait_async0();                   // next tile fully in LDS
    __syncthreads();                 // everyone done reading current buffer
  }

  // ---- epilogue ----
  const int lr = lane & 15;
  const int lm = (lane >> 4) * 8;
  #pragma unroll
  for (int mt = 0; mt < 2; mt++) {
    #pragma unroll
    for (int nt = 0; nt < 4; nt++) {
      const int col = bN + waveN * 64 + nt * 16 + lr;
      const float bv = (bias != nullptr && col < Ncols) ? bias[col] : 0.0f;
      #pragma unroll
      for (int r = 0; r < 8; r++) {
        const int row = bM + waveM * 32 + mt * 16 + lm + r;
        float v = acc[mt][nt][r] + bv;
        if constexpr (EPI == 0) {
          Cf[(size_t)row * ldC + col] = v;
        } else if constexpr (EPI == 1) {
          const float g = 1.0f / (1.0f + expf(-v)) + 0.1f;  // gate + 0.1 skip
          const size_t idx = (size_t)row * ldC + col;
          supR[idx] *= g;
          supI[idx] *= g;
        } else if constexpr (EPI == 2) {
          const float ge = 0.5f * v * (1.0f + erff(v * 0.70710678118654752f));
          Cb[(size_t)row * ldC + col] = f2bf(ge);
        } else {
          if (col < Ncols) Cf[(size_t)row * ldC + col] = v;
        }
      }
    }
  }
}

// ---------------------------------------------------------------------------
// Block-wide sum reduction, 256 threads
// ---------------------------------------------------------------------------
__device__ __forceinline__ float blockReduceSum256(float v, float* red) {
  const int tid = threadIdx.x;
  red[tid] = v;
  __syncthreads();
  #pragma unroll
  for (int s = 128; s > 0; s >>= 1) {
    if (tid < s) red[tid] += red[tid + s];
    __syncthreads();
  }
  const float r = red[0];
  __syncthreads();
  return r;
}

// ---------------------------------------------------------------------------
// h = gelu(layernorm(y, g, b))  per row of 1024; writes bf16 for next GEMM
// ---------------------------------------------------------------------------
__global__ __launch_bounds__(256)
void ln_gelu_kernel(const float* __restrict__ y, const float* __restrict__ g,
                    const float* __restrict__ b, __bf16* __restrict__ h) {
  __shared__ float red[256];
  const int row = blockIdx.x, tid = threadIdx.x;
  const float* yr = y + (size_t)row * 1024;
  float x[4];
  float s = 0.0f;
  #pragma unroll
  for (int i = 0; i < 4; i++) { x[i] = yr[tid + i * 256]; s += x[i]; }
  const float mu = blockReduceSum256(s, red) * (1.0f / 1024.0f);
  float q = 0.0f;
  #pragma unroll
  for (int i = 0; i < 4; i++) { const float d = x[i] - mu; q += d * d; }
  const float var = blockReduceSum256(q, red) * (1.0f / 1024.0f);
  const float inv = rsqrtf(var + 1e-5f);
  #pragma unroll
  for (int i = 0; i < 4; i++) {
    const int e = tid + i * 256;
    const float t = (x[i] - mu) * inv * g[e] + b[e];
    const float ge = 0.5f * t * (1.0f + erff(t * 0.70710678118654752f));
    h[(size_t)row * 1024 + e] = f2bf(ge);
  }
}

// ---------------------------------------------------------------------------
// Wave kernel: per batch row, for s=0..2: LN(2048) -> exp(-t^2) -> partial-norm
// rescale -> phase rotation; then cross-s cosine-sim softmax -> superposition.
// ---------------------------------------------------------------------------
__global__ __launch_bounds__(256)
void wave_kernel(const float* __restrict__ out2, const float* __restrict__ ln_g,
                 const float* __restrict__ ln_b, const float* __restrict__ phases,
                 const float* __restrict__ temperature,
                 float* __restrict__ supR, float* __restrict__ supI) {
  __shared__ float red[256];
  __shared__ float wr[3][1024];
  __shared__ float wi[3][1024];
  __shared__ float gbuf[2048];
  const int bidx = blockIdx.x, tid = threadIdx.x;

  for (int s = 0; s < 3; s++) {
    const float* rowp = out2 + (size_t)bidx * 6144 + s * 2048;
    float x[8];
    float sum = 0.0f;
    #pragma unroll
    for (int i = 0; i < 8; i++) { x[i] = rowp[tid + i * 256]; sum += x[i]; }
    const float mu = blockReduceSum256(sum, red) * (1.0f / 2048.0f);
    float q = 0.0f;
    #pragma unroll
    for (int i = 0; i < 8; i++) { const float d = x[i] - mu; q += d * d; }
    const float var = blockReduceSum256(q, red) * (1.0f / 2048.0f);
    const float inv = rsqrtf(var + 1e-5f);
    float ns = 0.0f;
    #pragma unroll
    for (int i = 0; i < 8; i++) {
      const int e = tid + i * 256;
      const float t = (x[i] - mu) * inv * ln_g[s * 2048 + e] + ln_b[s * 2048 + e];
      const float gg = expf(-t * t);
      gbuf[e] = gg;
      ns += gg * gg;
    }
    ns = blockReduceSum256(ns, red) + EPSV;       // barrier makes gbuf visible
    const float factor = sqrtf(1.5f * 1.5f / ns); // PARTIAL_NORM = 1.5
    const float ph = phases[s];
    const float c = cosf(ph), sn = sinf(ph);
    #pragma unroll
    for (int i = 0; i < 4; i++) {
      const int hh = tid + i * 256;
      const float a  = gbuf[hh] * factor;
      const float be = gbuf[1024 + hh] * factor;
      wr[s][hh] = a * c - be * sn;
      wi[s][hh] = a * sn + be * c;
    }
    __syncthreads();
  }

  // mean over s into gbuf[0..1023]
  #pragma unroll
  for (int i = 0; i < 4; i++) {
    const int hh = tid + i * 256;
    gbuf[hh] = (wr[0][hh] + wr[1][hh] + wr[2][hh]) * (1.0f / 3.0f);
  }
  __syncthreads();
  float mn = 0.0f;
  #pragma unroll
  for (int i = 0; i < 4; i++) { const float m = gbuf[tid + i * 256]; mn += m * m; }
  const float mean_norm = sqrtf(blockReduceSum256(mn, red)) + EPSV;

  float cosv[3];
  for (int s = 0; s < 3; s++) {
    float wn = 0.0f, dot = 0.0f;
    #pragma unroll
    for (int i = 0; i < 4; i++) {
      const int hh = tid + i * 256;
      const float w = wr[s][hh];
      wn += w * w;
      dot += w * gbuf[hh];
    }
    wn = blockReduceSum256(wn, red);
    dot = blockReduceSum256(dot, red);
    cosv[s] = dot / ((sqrtf(wn) + EPSV) * mean_norm);
  }
  const float invT = 1.0f / temperature[0];
  float a0 = cosv[0] * invT, a1 = cosv[1] * invT, a2 = cosv[2] * invT;
  const float mx = fmaxf(a0, fmaxf(a1, a2));
  const float e0 = expf(a0 - mx), e1 = expf(a1 - mx), e2 = expf(a2 - mx);
  const float iz = 1.0f / (e0 + e1 + e2);
  const float w0 = e0 * iz, w1 = e1 * iz, w2 = e2 * iz;
  #pragma unroll
  for (int i = 0; i < 4; i++) {
    const int hh = tid + i * 256;
    const size_t idx = (size_t)bidx * 1024 + hh;
    supR[idx] = w0 * wr[0][hh] + w1 * wr[1][hh] + w2 * wr[2][hh];
    supI[idx] = w0 * wi[0][hh] + w1 * wi[1][hh] + w2 * wi[2][hh];
  }
}

// ---------------------------------------------------------------------------
// mag = sqrt(supR^2 + supI^2 + eps) -> bf16 (input to the gate GEMM)
// ---------------------------------------------------------------------------
__global__ void mag_kernel(const float* __restrict__ r, const float* __restrict__ im,
                           __bf16* __restrict__ mag, long n) {
  long i = (long)blockIdx.x * blockDim.x + threadIdx.x;
  const long st = (long)gridDim.x * blockDim.x;
  for (; i < n; i += st) {
    const float a = r[i], b = im[i];
    mag[i] = f2bf(sqrtf(a * a + b * b + EPSV));
  }
}

// ---------------------------------------------------------------------------
// Top-8 of mag_sq per row, mask + normalize -> sparse probs row (bf16)
// ---------------------------------------------------------------------------
__global__ __launch_bounds__(256)
void topk_probs_kernel(const float* __restrict__ supR, const float* __restrict__ supI,
                       __bf16* __restrict__ probs) {
  __shared__ float vals[1024];
  __shared__ float rv[256];
  __shared__ int   ri[256];
  __shared__ float pv[8];
  __shared__ int   pi[8];
  const int bidx = blockIdx.x, tid = threadIdx.x;

  #pragma unroll
  for (int i = 0; i < 4; i++) {
    const int hh = tid + i * 256;
    const float r = supR[(size_t)bidx * 1024 + hh];
    const float im = supI[(size_t)bidx * 1024 + hh];
    vals[hh] = r * r + im * im;
  }
  __syncthreads();

  for (int k = 0; k < 8; k++) {
    float bv = -1.0f; int bi = -1;
    #pragma unroll
    for (int i = 0; i < 4; i++) {
      const int hh = tid + i * 256;
      const float v = vals[hh];
      if (v > bv) { bv = v; bi = hh; }
    }
    rv[tid] = bv; ri[tid] = bi;
    __syncthreads();
    #pragma unroll
    for (int s = 128; s > 0; s >>= 1) {
      if (tid < s && rv[tid + s] > rv[tid]) { rv[tid] = rv[tid + s]; ri[tid] = ri[tid + s]; }
      __syncthreads();
    }
    if (tid == 0) { pv[k] = rv[0]; pi[k] = ri[0]; vals[ri[0]] = -2.0f; }
    __syncthreads();
  }

  float tot = EPSV;
  #pragma unroll
  for (int k = 0; k < 8; k++) tot += pv[k];
  const float itot = 1.0f / tot;

  #pragma unroll
  for (int i = 0; i < 4; i++)
    probs[(size_t)bidx * 1024 + tid + i * 256] = (__bf16)0.0f;
  __syncthreads();
  if (tid < 8)
    probs[(size_t)bidx * 1024 + pi[tid]] = f2bf(pv[tid] * itot);
}

// ---------------------------------------------------------------------------
// Host-side orchestration
// ---------------------------------------------------------------------------
extern "C" void kernel_launch(void* const* d_in, const int* in_sizes, int n_in,
                              void* d_out, int out_size, void* d_ws, size_t ws_size,
                              hipStream_t stream) {
  (void)in_sizes; (void)n_in; (void)out_size; (void)ws_size;

  const float* x      = (const float*)d_in[0];
  const float* proj_w = (const float*)d_in[1];
  const float* proj_b = (const float*)d_in[2];
  const float* ln_p_g = (const float*)d_in[3];
  const float* ln_p_b = (const float*)d_in[4];
  const float* wf_w   = (const float*)d_in[5];
  const float* wf_b   = (const float*)d_in[6];
  const float* ln_g   = (const float*)d_in[7];
  const float* ln_b   = (const float*)d_in[8];
  const float* temp   = (const float*)d_in[9];
  const float* phases = (const float*)d_in[10];
  const float* gate_w = (const float*)d_in[11];
  const float* gate_b = (const float*)d_in[12];
  const float* cls_w1 = (const float*)d_in[13];
  const float* cls_b1 = (const float*)d_in[14];
  const float* cls_w2 = (const float*)d_in[15];
  const float* cls_b2 = (const float*)d_in[16];
  float* out = (float*)d_out;

  const int B = 8192, D = 2048, P = 1024, H = 1024, S = 3;
  const int N2 = S * 2 * H;     // 6144
  const int C = 1000, CP = 1024;

  char* ws = (char*)d_ws;
  size_t off = 0;
  auto take = [&](size_t bytes) -> char* {
    char* p = ws + off;
    off = (off + bytes + 255) & ~(size_t)255;
    return p;
  };
  __bf16* xB     = (__bf16*)take((size_t)B * D * 2);
  __bf16* wProj  = (__bf16*)take((size_t)P * D * 2);
  __bf16* wWf    = (__bf16*)take((size_t)N2 * P * 2);
  __bf16* wGate  = (__bf16*)take((size_t)H * H * 2);
  __bf16* wCls1  = (__bf16*)take((size_t)H * H * 2);
  __bf16* wCls2  = (__bf16*)take((size_t)CP * H * 2);
  float*  y1     = (float*)take((size_t)B * P * 4);
  __bf16* hB     = (__bf16*)take((size_t)B * P * 2);
  float*  out2   = (float*)take((size_t)B * N2 * 4);
  float*  supR   = (float*)take((size_t)B * H * 4);
  float*  supI   = (float*)take((size_t)B * H * 4);
  __bf16* magB   = (__bf16*)take((size_t)B * H * 2);
  __bf16* probsB = (__bf16*)take((size_t)B * H * 2);
  __bf16* hidB   = (__bf16*)take((size_t)B * H * 2);

  auto cvt = [&](const float* s, __bf16* dptr, long n, long nd) {
    const int blocks = (int)((nd + 1023) / 1024);
    cvt_f32_bf16<<<blocks, 256, 0, stream>>>(s, dptr, n, nd);
  };
  cvt(x,      xB,    (long)B * D, (long)B * D);    // one pass: halves A traffic
  cvt(proj_w, wProj, (long)P * D, (long)P * D);
  cvt(wf_w,   wWf,   (long)N2 * P, (long)N2 * P);
  cvt(gate_w, wGate, (long)H * H, (long)H * H);
  cvt(cls_w1, wCls1, (long)H * H, (long)H * H);
  cvt(cls_w2, wCls2, (long)C * H, (long)CP * H);   // zero-pad rows 1000..1023

  // 1) y1 = x @ proj_w^T + proj_b          (8192x1024, K=2048)
  gemm_wmma<0><<<dim3(P / 128, B / 128), 256, 0, stream>>>(
      xB, wProj, y1, nullptr, proj_b, nullptr, nullptr,
      B, P, D, P, P);

  // 2) h = gelu(LN(y1)) -> bf16
  ln_gelu_kernel<<<B, 256, 0, stream>>>(y1, ln_p_g, ln_p_b, hB);

  // 3) out2 = h @ wf_w^T + wf_b            (8192x6144, K=1024)
  gemm_wmma<0><<<dim3(N2 / 128, B / 128), 256, 0, stream>>>(
      hB, wWf, out2, nullptr, wf_b, nullptr, nullptr,
      B, N2, P, N2, N2);

  // 4) LN / exp(-t^2) / rescale / rotate / cosine-softmax superposition
  wave_kernel<<<B, 256, 0, stream>>>(out2, ln_g, ln_b, phases, temp, supR, supI);

  // 5) two self-modulation steps: mag -> gate GEMM with sigmoid RMW epilogue
  for (int it = 0; it < 2; it++) {
    mag_kernel<<<2048, 256, 0, stream>>>(supR, supI, magB, (long)B * H);
    gemm_wmma<1><<<dim3(H / 128, B / 128), 256, 0, stream>>>(
        magB, wGate, nullptr, nullptr, gate_b, supR, supI,
        B, H, H, H, H);
  }

  // 6) top-8 mask + normalize -> sparse probs (bf16)
  topk_probs_kernel<<<B, 256, 0, stream>>>(supR, supI, probsB);

  // 7) hid = gelu(probs @ cls_w1^T + cls_b1) -> bf16
  gemm_wmma<2><<<dim3(H / 128, B / 128), 256, 0, stream>>>(
      probsB, wCls1, nullptr, hidB, cls_b1, nullptr, nullptr,
      B, H, H, H, H);

  // 8) logits = hid @ cls_w2^T + cls_b2  (N padded 1024, store cols < 1000)
  gemm_wmma<3><<<dim3(CP / 128, B / 128), 256, 0, stream>>>(
      hidB, wCls2, out, nullptr, cls_b2, nullptr, nullptr,
      B, CP, H, C, C);
}
